// AttentionTSP_2199023255584
// MI455X (gfx1250) — matrix operand order repaired and checked by reading, not compiled, not run on hardware
//
#include <hip/hip_runtime.h>
#include <hip/hip_bf16.h>
#include <cstdint>

typedef __attribute__((ext_vector_type(16))) _Float16 v16h;
typedef __attribute__((ext_vector_type(8)))  float    v8f;

#define Bb 1024
#define Ss 256
#define Dd 128
#define FFd 2048
#define BS (Bb * Ss)

// ---------------------------------------------------------------------------
// WMMA helpers (CDNA5 16x16x32 f16 -> f32)
// ---------------------------------------------------------------------------
__device__ __forceinline__ v8f wmma16(v16h a, v16h b, v8f c) {
    return __builtin_amdgcn_wmma_f32_16x16x32_f16(false, a, false, b, (short)0, c,
                                                  false, false);
}

// A-fragment (16x32, 16-bit): lanes 0-15 row M=lane, K={0..7,16..23};
// lanes 16-31 row M=lane-16, K={8..15,24..31}.
__device__ __forceinline__ v16h load_a_f16(const _Float16* A, int lda, int row0,
                                           int k0, int lane) {
    const int hi = (lane >> 4) & 1;
    const _Float16* p = A + (size_t)(row0 + (lane & 15)) * lda;
    const int kb1 = k0 + hi * 8;
    const int kb2 = k0 + 16 + hi * 8;
    v16h f;
#pragma unroll
    for (int j = 0; j < 8; ++j) f[j] = p[kb1 + j];
#pragma unroll
    for (int j = 0; j < 8; ++j) f[8 + j] = p[kb2 + j];
    return f;
}

__device__ __forceinline__ v16h load_a_f32cvt(const float* A, int lda, int row0,
                                              int k0, int lane) {
    const int hi = (lane >> 4) & 1;
    const float* p = A + (size_t)(row0 + (lane & 15)) * lda;
    const int kb1 = k0 + hi * 8;
    const int kb2 = k0 + 16 + hi * 8;
    v16h f;
#pragma unroll
    for (int j = 0; j < 8; ++j) f[j] = (_Float16)p[kb1 + j];
#pragma unroll
    for (int j = 0; j < 8; ++j) f[8 + j] = (_Float16)p[kb2 + j];
    return f;
}

// B-fragment (32x16, 16-bit): lanes 0-15 col N=lane hold K={0..7,16..23}; lanes 16-31
// col N=lane-16 hold K={8..15,24..31}.  Source is row-major [N, K] so each lane reads
// two runs of 8 contiguous halves.
__device__ __forceinline__ v16h load_b_f16(const _Float16* Bw, int ldb, int col0,
                                           int k0, int lane) {
    const int hi = (lane >> 4) & 1;
    const _Float16* p = Bw + (size_t)(col0 + (lane & 15)) * ldb + k0 + hi * 8;
    v16h f;
#pragma unroll
    for (int j = 0; j < 8; ++j) f[j] = p[j];
#pragma unroll
    for (int j = 0; j < 8; ++j) f[8 + j] = p[16 + j];
    return f;
}

// Low 32 bits of a generic shared-memory address are the LDS byte offset
// (ISA 10.2, LDS aperture).
__device__ __forceinline__ unsigned lds_off_u32(const void* p) {
    return (unsigned)(uintptr_t)p;
}

// ---------------------------------------------------------------------------
// Generic WMMA GEMM:  C[M,N] = A[M,K] * Bw[N,K]^T  (+ epilogue)
// block = 256 threads = 8 waves; block tile 128x64; wave tile 32x32.
// B tiles (shared by all 4 wave-rows) are staged into LDS with the CDNA5
// async copy path (global_load_async_to_lds_b128 / ASYNCcnt), double buffered.
// ---------------------------------------------------------------------------
enum { EPI_QKV = 0, EPI_RELU16 = 1, EPI_ADDRES = 2, EPI_BIAS32 = 3 };

template <int EPI, bool AF32>
__global__ __launch_bounds__(256, 2) void gemm_wmma(
        const void* __restrict__ Ap,
        const _Float16* __restrict__ Bw,
        const float* __restrict__ bias,
        const float* __restrict__ res,
        void* __restrict__ out0,
        _Float16* __restrict__ out1,
        int M, int N, int K) {
    __shared__ _Float16 btile[2][64 * 32];  // [buf][col (64)][k (32)]

    const int tid  = threadIdx.x;
    const int wave = tid >> 5;
    const int lane = tid & 31;
    const int wr = wave & 3;   // 4 wave rows of 32
    const int wc = wave >> 2;  // 2 wave cols of 32
    const int row0 = blockIdx.x * 128 + wr * 32;
    const int col0_blk = blockIdx.y * 64;
    const int wcol = wc * 32;  // wave col offset inside the 64-wide block tile

    // async-stage one 64x32 f16 B tile into LDS (256 threads x 16B = 4KB)
    auto stage = [&](int k0, int buf) {
        const int c  = tid >> 2;  // 0..63 : column within tile
        const int ch = tid & 3;   // 0..3  : 8-half chunk within the 32-k row
        const _Float16* g = Bw + (size_t)(col0_blk + c) * K + k0 + ch * 8;
        unsigned dst = lds_off_u32(&btile[buf][c * 32 + ch * 8]);
        asm volatile("global_load_async_to_lds_b128 %0, %1, off"
                     :: "v"(dst), "v"(g) : "memory");
    };

    stage(0, 0);
    int buf = 0;
    v8f acc[2][2] = {};
    for (int k0 = 0; k0 < K; k0 += 32) {
        asm volatile("s_wait_asynccnt 0" ::: "memory");
        __syncthreads();
        if (k0 + 32 < K) stage(k0 + 32, buf ^ 1);

        v16h a0, a1;
        if constexpr (AF32) {
            const float* A = (const float*)Ap;
            if (k0 + 32 < K)
                __builtin_prefetch(A + (size_t)(row0 + (lane & 15)) * K + k0 + 32, 0, 0);
            a0 = load_a_f32cvt(A, K, row0, k0, lane);
            a1 = load_a_f32cvt(A, K, row0 + 16, k0, lane);
        } else {
            const _Float16* A = (const _Float16*)Ap;
            if (k0 + 32 < K)
                __builtin_prefetch(A + (size_t)(row0 + (lane & 15)) * K + k0 + 32, 0, 0);
            a0 = load_a_f16(A, K, row0, k0, lane);
            a1 = load_a_f16(A, K, row0 + 16, k0, lane);
        }
        v16h b0 = load_b_f16(btile[buf], 32, wcol, 0, lane);
        v16h b1 = load_b_f16(btile[buf], 32, wcol + 16, 0, lane);
        acc[0][0] = wmma16(a0, b0, acc[0][0]);
        acc[0][1] = wmma16(a0, b1, acc[0][1]);
        acc[1][0] = wmma16(a1, b0, acc[1][0]);
        acc[1][1] = wmma16(a1, b1, acc[1][1]);
        buf ^= 1;
    }

    // C layout: lanes 0-15 -> (M=r, N=lane); lanes 16-31 -> (M=8+r, N=lane-16)
    const int hi = (lane >> 4) & 1;
    const int ln = lane & 15;
#pragma unroll
    for (int i = 0; i < 2; ++i)
#pragma unroll
        for (int j = 0; j < 2; ++j) {
            const int mb = row0 + i * 16 + hi * 8;
            const int n  = col0_blk + wcol + j * 16 + ln;
#pragma unroll
            for (int r = 0; r < 8; ++r) {
                const int m = mb + r;
                float v = acc[i][j][r];
                if (EPI == EPI_QKV) {
                    v += bias[n];
                    if (n < 256) {
                        ((_Float16*)out0)[(size_t)m * 256 + n] = (_Float16)v;
                    } else {
                        const int bb = m >> 8, ss = m & 255;  // S == 256
                        out1[((size_t)bb * Dd + (n - 256)) * Ss + ss] = (_Float16)v;
                    }
                } else if (EPI == EPI_RELU16) {
                    v += bias[n];
                    v = fmaxf(v, 0.0f);
                    ((_Float16*)out0)[(size_t)m * N + n] = (_Float16)v;
                } else if (EPI == EPI_ADDRES) {
                    v += bias[n] + res[(size_t)m * N + n];
                    ((float*)out0)[(size_t)m * N + n] = v;
                } else {
                    v += bias[n];
                    ((float*)out0)[(size_t)m * N + n] = v;
                }
            }
        }
}

// ---------------------------------------------------------------------------
// Attention: per (batch, 64-row q block), softmax(QK^T/sqrt(D)) V with WMMA.
// block = 128 threads (4 waves); wave handles 16 query rows x all 256 keys.
// launch_bounds(128,1): keep all 16 score tiles in registers (no spills).
// ---------------------------------------------------------------------------
__global__ __launch_bounds__(128, 1) void attn_kernel(
        const _Float16* __restrict__ qk,
        const _Float16* __restrict__ vT,
        _Float16* __restrict__ ao) {
    __shared__ _Float16 plds[4 * 16 * 256];  // probs per wave, row-major [16,256]

    const int b    = blockIdx.x;
    const int qblk = blockIdx.y;
    const int wave = threadIdx.x >> 5;
    const int lane = threadIdx.x & 31;
    const int hi = (lane >> 4) & 1;
    const int ln = lane & 15;
    const int qr = qblk * 64 + wave * 16;

    const _Float16* qbase = qk + (size_t)b * Ss * 256;        // Q at cols [0,128)
    const _Float16* kbase = qk + (size_t)b * Ss * 256 + 128;  // K at cols [128,256)

    // Preload 4 A-fragments of Q (K = 0..127)
    v16h aq[4];
#pragma unroll
    for (int kt = 0; kt < 4; ++kt) aq[kt] = load_a_f16(qbase, 256, qr, kt * 32, lane);

    // Scores: 16 n-tiles over the 256 keys
    v8f sc[16];
#pragma unroll
    for (int nt = 0; nt < 16; ++nt) {
        v8f c = {};
#pragma unroll
        for (int kt = 0; kt < 4; ++kt) {
            v16h bk = load_b_f16(kbase, 256, nt * 16, kt * 32, lane);
            c = wmma16(aq[kt], bk, c);
        }
        sc[nt] = c;
    }

    const float scale = 0.08838834764831845f;  // 1/sqrt(128)
    float inv_sum[8];
#pragma unroll
    for (int r = 0; r < 8; ++r) {
        float mx = -1e30f;
#pragma unroll
        for (int nt = 0; nt < 16; ++nt) mx = fmaxf(mx, sc[nt][r]);
        for (int m = 1; m < 16; m <<= 1) mx = fmaxf(mx, __shfl_xor(mx, m, 32));
        float s = 0.0f;
#pragma unroll
        for (int nt = 0; nt < 16; ++nt) {
            float p = expf(scale * (sc[nt][r] - mx));
            sc[nt][r] = p;
            s += p;
        }
        for (int m = 1; m < 16; m <<= 1) s += __shfl_xor(s, m, 32);
        inv_sum[r] = 1.0f / s;
    }

    // write probs to LDS in row-major [16,256] (per wave)
    _Float16* pl = plds + wave * (16 * 256);
#pragma unroll
    for (int nt = 0; nt < 16; ++nt)
#pragma unroll
        for (int r = 0; r < 8; ++r)
            pl[(size_t)(r + hi * 8) * 256 + nt * 16 + ln] =
                (_Float16)(sc[nt][r] * inv_sum[r]);
    __syncthreads();

    // P @ V : A from LDS probs, B from pre-transposed V (contiguous per lane)
    v16h pa[8];
#pragma unroll
    for (int kt = 0; kt < 8; ++kt) pa[kt] = load_a_f16(pl, 256, 0, kt * 32, lane);

    const _Float16* vb = vT + (size_t)b * Dd * Ss;
    const size_t rowbase = (size_t)b * Ss + qr;
#pragma unroll
    for (int dt = 0; dt < 8; ++dt) {
        v8f c = {};
#pragma unroll
        for (int kt = 0; kt < 8; ++kt) {
            v16h bv = load_b_f16(vb, 256, dt * 16, kt * 32, lane);
            c = wmma16(pa[kt], bv, c);
        }
#pragma unroll
        for (int r = 0; r < 8; ++r)
            ao[(rowbase + hi * 8 + r) * Dd + dt * 16 + ln] = (_Float16)c[r];
    }
}

// ---------------------------------------------------------------------------
// Elementwise / small kernels
// ---------------------------------------------------------------------------
__global__ void cvt_f16_kernel(const float* __restrict__ src,
                               _Float16* __restrict__ dst, int n) {
    int i = blockIdx.x * 256 + threadIdx.x;
    if (i < n) dst[i] = (_Float16)src[i];
}

__global__ void embed_kernel(const float* __restrict__ inp,
                             const float* __restrict__ ew,
                             float* __restrict__ emb) {
    size_t idx = (size_t)blockIdx.x * 256 + threadIdx.x;  // over BS*D
    size_t nrow = idx >> 7;
    int d = (int)(idx & 127);
    emb[idx] = inp[nrow * 2] * ew[d] + inp[nrow * 2 + 1] * ew[Dd + d];
}

__global__ void ln_kernel(const float* __restrict__ in, const float* __restrict__ g,
                          const float* __restrict__ be, float* __restrict__ out) {
    const int wave = threadIdx.x >> 5, lane = threadIdx.x & 31;
    const size_t row = (size_t)blockIdx.x * 8 + wave;
    const float* p = in + row * Dd;
    float x[4];
    float s = 0.0f;
#pragma unroll
    for (int j = 0; j < 4; ++j) {
        x[j] = p[lane * 4 + j];
        s += x[j];
    }
    for (int m = 1; m < 32; m <<= 1) s += __shfl_xor(s, m, 32);
    const float mean = s * (1.0f / 128.0f);
    float vs = 0.0f;
#pragma unroll
    for (int j = 0; j < 4; ++j) {
        float d = x[j] - mean;
        vs += d * d;
    }
    for (int m = 1; m < 32; m <<= 1) vs += __shfl_xor(vs, m, 32);
    const float rst = rsqrtf(vs * (1.0f / 128.0f) + 1e-5f);
#pragma unroll
    for (int j = 0; j < 4; ++j) {
        int d = lane * 4 + j;
        out[row * Dd + d] = (x[j] - mean) * rst * g[d] + be[d];
    }
}

__global__ void hbar_kernel(const float* __restrict__ h,
                            const float* __restrict__ hctx_w,
                            const float* __restrict__ hctx_b,
                            float* __restrict__ h_bar) {
    __shared__ float mbuf[Dd];
    const int b = blockIdx.x, d = threadIdx.x;
    float s = 0.0f;
    for (int t = 0; t < Ss; ++t) s += h[((size_t)b * Ss + t) * Dd + d];
    mbuf[d] = s * (1.0f / 256.0f);
    __syncthreads();
    float a = hctx_b[d];
    for (int k = 0; k < Dd; ++k) a += hctx_w[d * Dd + k] * mbuf[k];
    h_bar[(size_t)b * Dd + d] = a;
}

__global__ void vinit_kernel(const float* __restrict__ init_w,
                             const float* __restrict__ vwe_w,
                             const float* __restrict__ vwe_b,
                             float* __restrict__ vinit) {
    int j = threadIdx.x;  // 128 threads
    float a = vwe_b[j];
    for (int i = 0; i < 2 * Dd; ++i) a += vwe_w[j * 2 * Dd + i] * init_w[i];
    vinit[j] = a;
}

__global__ void qinit_kernel(const float* __restrict__ h_bar,
                             const float* __restrict__ vinit,
                             float* __restrict__ query,
                             unsigned char* __restrict__ mask) {
    const int b = blockIdx.x, t = threadIdx.x;  // 256 threads
    if (t < Dd) query[(size_t)b * Dd + t] = h_bar[(size_t)b * Dd + t] + vinit[t];
    mask[(size_t)b * Ss + t] = 0;
}

__device__ __forceinline__ unsigned hashu(unsigned x) {
    x ^= x >> 17; x *= 0xed5ad4bbu;
    x ^= x >> 11; x *= 0xac4c1b51u;
    x ^= x >> 15; x *= 0x31848babu;
    x ^= x >> 14;
    return x;
}

__global__ void decode_step(int step,
                            const float* __restrict__ ref,
                            const float* __restrict__ pv,
                            const float* __restrict__ pq_w,
                            const float* __restrict__ pq_b,
                            const float* __restrict__ h,
                            const float* __restrict__ h_bar,
                            const float* __restrict__ vwe_w,
                            const float* __restrict__ vwe_b,
                            float* __restrict__ query,
                            float* __restrict__ first_hs,
                            unsigned char* __restrict__ mask,
                            float* __restrict__ out) {
    __shared__ float qsh[Dd], qp[Dd], lg[Ss], red[Ss];
    __shared__ int redi[Ss];
    __shared__ float chsh[Dd], fsh[Dd];
    __shared__ int sh_ch;

    const int b = blockIdx.x, t = threadIdx.x;  // 256 threads

    if (t < Dd) qsh[t] = query[(size_t)b * Dd + t];
    __syncthreads();
    if (t < Dd) {
        float a = pq_b[t];
        for (int k = 0; k < Dd; ++k) a += pq_w[t * Dd + k] * qsh[k];
        qp[t] = a;
    }
    __syncthreads();

    // pointer logits: u = pv . tanh(ref_proj + qp)
    const float* rr = ref + ((size_t)b * Ss + t) * Dd;
    float u = 0.0f;
    for (int d = 0; d < Dd; ++d) u += pv[d] * tanhf(rr[d] + qp[d]);
    const float logit = mask[(size_t)b * Ss + t] ? -100000.0f : 10.0f * tanhf(u);
    lg[t] = logit;

    // max-reduce for logsumexp
    red[t] = logit;
    __syncthreads();
    for (int st = 128; st > 0; st >>= 1) {
        if (t < st) red[t] = fmaxf(red[t], red[t + st]);
        __syncthreads();
    }
    const float mx = red[0];
    __syncthreads();
    red[t] = expf(logit - mx);
    __syncthreads();
    for (int st = 128; st > 0; st >>= 1) {
        if (t < st) red[t] += red[t + st];
        __syncthreads();
    }
    const float lse = mx + logf(red[0]);
    __syncthreads();

    // Gumbel-max categorical sample
    unsigned hv = hashu(hashu((unsigned)step * 1664525u + (unsigned)b) ^
                        (0x9e3779b9u + (unsigned)t * 747796405u));
    float u01 = ((hv >> 8) + 0.5f) * (1.0f / 16777216.0f);
    float key = logit - logf(-logf(u01));
    red[t] = key;
    redi[t] = t;
    __syncthreads();
    for (int st = 128; st > 0; st >>= 1) {
        if (t < st && red[t + st] > red[t]) {
            red[t] = red[t + st];
            redi[t] = redi[t + st];
        }
        __syncthreads();
    }
    if (t == 0) {
        const int ch = redi[0];
        sh_ch = ch;
        out[(size_t)b * Ss + step] = lg[ch] - lse;            // logprobs
        out[(size_t)BS + (size_t)b * Ss + step] = (float)ch;  // indices
        mask[(size_t)b * Ss + ch] = 1;
    }
    __syncthreads();
    const int ch = sh_ch;

    // next query = h_bar + concat(first_hs, chs) @ vwe_w^T + vwe_b
    if (t < Dd) {
        float c = h[((size_t)b * Ss + ch) * Dd + t];
        chsh[t] = c;
        if (step == 0) {
            first_hs[(size_t)b * Dd + t] = c;
            fsh[t] = c;
        } else {
            fsh[t] = first_hs[(size_t)b * Dd + t];
        }
    }
    __syncthreads();
    if (t < Dd) {
        float a = vwe_b[t] + h_bar[(size_t)b * Dd + t];
        for (int i = 0; i < Dd; ++i)
            a += vwe_w[t * 2 * Dd + i] * fsh[i] + vwe_w[t * 2 * Dd + Dd + i] * chsh[i];
        query[(size_t)b * Dd + t] = a;
    }
}

// ---------------------------------------------------------------------------
// Host orchestration
// ---------------------------------------------------------------------------
extern "C" void kernel_launch(void* const* d_in, const int* in_sizes, int n_in,
                              void* d_out, int out_size, void* d_ws, size_t ws_size,
                              hipStream_t stream) {
    (void)in_sizes; (void)n_in; (void)out_size; (void)ws_size;
    const float* inputs    = (const float*)d_in[0];
    const float* embed_w   = (const float*)d_in[1];
    const float* in_proj_w = (const float*)d_in[2];
    const float* in_proj_b = (const float*)d_in[3];
    const float* out_proj_w= (const float*)d_in[4];
    const float* out_proj_b= (const float*)d_in[5];
    const float* ln1_g     = (const float*)d_in[6];
    const float* ln1_b     = (const float*)d_in[7];
    const float* ff1_w     = (const float*)d_in[8];
    const float* ff1_b     = (const float*)d_in[9];
    const float* ff2_w     = (const float*)d_in[10];
    const float* ff2_b     = (const float*)d_in[11];
    const float* ln2_g     = (const float*)d_in[12];
    const float* ln2_b     = (const float*)d_in[13];
    const float* init_w    = (const float*)d_in[14];
    const float* hctx_w    = (const float*)d_in[15];
    const float* hctx_b    = (const float*)d_in[16];
    const float* vwe_w     = (const float*)d_in[17];
    const float* vwe_b     = (const float*)d_in[18];
    const float* pq_w      = (const float*)d_in[19];
    const float* pq_b      = (const float*)d_in[20];
    const float* pr_w      = (const float*)d_in[21];
    const float* pr_b      = (const float*)d_in[22];
    const float* pv        = (const float*)d_in[23];
    float* out = (float*)d_out;

    char* ws = (char*)d_ws;
    // f16 weight pool
    _Float16* w16_in  = (_Float16*)(ws + 0);                 // 384*128
    _Float16* w16_out = w16_in + 384 * Dd;                   // 128*128
    _Float16* w16_ff1 = w16_out + Dd * Dd;                   // 2048*128
    _Float16* w16_ff2 = w16_ff1 + FFd * Dd;                  // 128*2048
    _Float16* w16_pr  = w16_ff2 + Dd * FFd;                  // 128*128
    size_t off = 2ull * 1024 * 1024;
    float*    bufEmb = (float*)(ws + off);        off += (size_t)BS * Dd * 4;   // reused: preLN1 / preLN2
    _Float16* bufQK  = (_Float16*)(ws + off);     size_t off_qk = off; off += (size_t)BS * 256 * 2;
    size_t off_vt = off;
    _Float16* bufVT  = (_Float16*)(ws + off);     off += (size_t)Bb * Dd * Ss * 2;
    _Float16* bufAO  = (_Float16*)(ws + off);     off += (size_t)BS * Dd * 2;
    float*    bufX   = (float*)(ws + off);        off += (size_t)BS * Dd * 4;
    float*    bufH   = (float*)(ws + off);        off += (size_t)BS * Dd * 4;
    float*    h_bar  = (float*)(ws + off);        off += (size_t)Bb * Dd * 4;
    float*    vinit  = (float*)(ws + off);        off += 512;
    float*    query  = (float*)(ws + off);        off += (size_t)Bb * Dd * 4;
    float*    firsth = (float*)(ws + off);        off += (size_t)Bb * Dd * 4;
    unsigned char* mask = (unsigned char*)(ws + off); off += (size_t)BS;
    float*    bufRef = (float*)(ws + off_qk);     // alias QK (dead after attention)
    _Float16* bufFF  = (_Float16*)(ws + off_vt);  // alias VT+AO (dead after out_proj)

    // 1) weights -> f16
    cvt_f16_kernel<<<(384 * Dd + 255) / 256, 256, 0, stream>>>(in_proj_w, w16_in, 384 * Dd);
    cvt_f16_kernel<<<(Dd * Dd + 255) / 256, 256, 0, stream>>>(out_proj_w, w16_out, Dd * Dd);
    cvt_f16_kernel<<<(FFd * Dd + 255) / 256, 256, 0, stream>>>(ff1_w, w16_ff1, FFd * Dd);
    cvt_f16_kernel<<<(Dd * FFd + 255) / 256, 256, 0, stream>>>(ff2_w, w16_ff2, Dd * FFd);
    cvt_f16_kernel<<<(Dd * Dd + 255) / 256, 256, 0, stream>>>(pr_w, w16_pr, Dd * Dd);

    // 2) embedding
    embed_kernel<<<(size_t)BS * Dd / 256, 256, 0, stream>>>(inputs, embed_w, bufEmb);

    // 3) QKV projection (writes Q/K f16 + V transposed)
    gemm_wmma<EPI_QKV, true><<<dim3(BS / 128, 384 / 64), 256, 0, stream>>>(
        bufEmb, w16_in, in_proj_b, nullptr, bufQK, bufVT, BS, 384, Dd);

    // 4) attention
    attn_kernel<<<dim3(Bb, 4), 128, 0, stream>>>(bufQK, bufVT, bufAO);

    // 5) out_proj + residual(emb) -> preLN1 (in place over bufEmb)
    gemm_wmma<EPI_ADDRES, false><<<dim3(BS / 128, Dd / 64), 256, 0, stream>>>(
        bufAO, w16_out, out_proj_b, bufEmb, bufEmb, nullptr, BS, Dd, Dd);

    // 6) LN1 -> x
    ln_kernel<<<BS / 8, 256, 0, stream>>>(bufEmb, ln1_g, ln1_b, bufX);

    // 7) FFN, chunked over rows (intermediate reuses workspace slice)
    const int CH = 32768;
    for (int c = 0; c < BS / CH; ++c) {
        const size_t ro = (size_t)c * CH * Dd;
        gemm_wmma<EPI_RELU16, true><<<dim3(CH / 128, FFd / 64), 256, 0, stream>>>(
            bufX + ro, w16_ff1, ff1_b, nullptr, bufFF, nullptr, CH, FFd, Dd);
        gemm_wmma<EPI_ADDRES, false><<<dim3(CH / 128, Dd / 64), 256, 0, stream>>>(
            bufFF, w16_ff2, ff2_b, bufX + ro, bufEmb + ro, nullptr, CH, Dd, FFd);
    }

    // 8) LN2 -> h
    ln_kernel<<<BS / 8, 256, 0, stream>>>(bufEmb, ln2_g, ln2_b, bufH);

    // 9) context vector
    hbar_kernel<<<Bb, Dd, 0, stream>>>(bufH, hctx_w, hctx_b, h_bar);

    // 10) ref_proj = h @ pr_w^T + pr_b
    gemm_wmma<EPI_BIAS32, true><<<dim3(BS / 128, Dd / 64), 256, 0, stream>>>(
        bufH, w16_pr, pr_b, nullptr, bufRef, nullptr, BS, Dd, Dd);

    // 11) decode init
    vinit_kernel<<<1, Dd, 0, stream>>>(init_w, vwe_w, vwe_b, vinit);
    qinit_kernel<<<Bb, Ss, 0, stream>>>(h_bar, vinit, query, mask);

    // 12) sequential pointer decode (256 steps)
    for (int step = 0; step < Ss; ++step) {
        decode_step<<<Bb, Ss, 0, stream>>>(step, bufRef, pv, pq_w, pq_b, bufH, h_bar,
                                           vwe_w, vwe_b, query, firsth, mask, out);
    }
}
